// Field2Prior_19121194402354
// MI455X (gfx1250) — compile-verified, matrix operand order checked by reading, not run on hardware
//
#include <hip/hip_runtime.h>
#include <hip/hip_bf16.h>
#include <math.h>

typedef __attribute__((ext_vector_type(16))) _Float16 v16h;
typedef __attribute__((ext_vector_type(2)))  __fp16   v2fp16;
typedef __attribute__((ext_vector_type(8)))  float    v8f;

constexpr int BATCH  = 1024;
constexpr int GRID   = 64;
constexpr int KWP    = 5;
constexpr int TSTEP  = 20;
constexpr int NPATH  = 15;   // NV * KWP
constexpr int MSEL   = 3;
constexpr int HIDDEN = 256;

union Frag { v16h v; unsigned u[8]; };

__device__ __forceinline__ unsigned pack2h(float a, float b)
{
#if __has_builtin(__builtin_amdgcn_cvt_pkrtz)
    union { v2fp16 h; unsigned u; } cv;
    cv.h = __builtin_amdgcn_cvt_pkrtz(a, b);
    return cv.u;
#else
    union { _Float16 h[2]; unsigned u; } cv;
    cv.h[0] = (_Float16)a; cv.h[1] = (_Float16)b; return cv.u;
#endif
}

// ---------------------------------------------------------------------------
// WMMA GEMM:  C = relu(A[MxK] @ W[KxN] + bias[N])
// Requires: M % 64 == 0, N % 32 == 0, K % 32 == 0 (all launches satisfy this).
// block = 128 threads (4 waves); block tile = 64 rows x 32 cols; K step 32.
// LDS double-buffered packed f16 pairs; one barrier per K step.
// ---------------------------------------------------------------------------
__global__ void gemm_relu_wmma(const float* __restrict__ A,
                               const float* __restrict__ W,
                               const float* __restrict__ bias,
                               float* __restrict__ C,
                               int M, int N, int K)
{
    __shared__ unsigned As[2][64][16];   // 64 rows x 16 k-pairs
    __shared__ unsigned Ws[2][32][16];   // 32 cols x 16 k-pairs
    const int tid  = threadIdx.x;
    const int lane = tid & 31;
    const int wave = tid >> 5;
    const int hh   = (lane >> 4) & 1;    // ISA lane-half for 16-bit operands
    const int ml   = lane & 15;
    const int m0   = blockIdx.y * 64;
    const int n0   = blockIdx.x * 32;

    v8f acc0 = {}, acc1 = {};
    const int nsteps = K >> 5;
    for (int s = 0; s < nsteps; ++s) {
        const int k0 = s << 5;
        const int p  = s & 1;
        // Stage A tile: 64x32 halves = 1024 pairs, 8 per thread, contiguous k.
#pragma unroll
        for (int it = 0; it < 8; ++it) {
            int idx = tid + it * 128;
            int r = idx >> 4, cp = idx & 15;
            const float* src = A + (size_t)(m0 + r) * K + k0 + 2 * cp;
            As[p][r][cp] = pack2h(src[0], src[1]);
            __builtin_prefetch(src + 32, 0, 1);   // speculative; safe past end
        }
        // Stage W tile: 32 cols x 16 pairs; pair elements are N apart.
#pragma unroll
        for (int it = 0; it < 4; ++it) {
            int idx = tid + it * 128;
            int cp = idx >> 5, n = idx & 31;
            const float* src = W + (size_t)(k0 + 2 * cp) * N + n0 + n;
            Ws[p][n][cp] = pack2h(src[0], src[N]);
        }
        __syncthreads();
        Frag a, b0, b1;
#pragma unroll
        for (int j = 0; j < 8; ++j) {
            int kp = ((j < 4) ? j : (8 + (j - 4))) + hh * 4;
            a.u[j]  = As[p][wave * 16 + ml][kp];
            b0.u[j] = Ws[p][ml][kp];
            b1.u[j] = Ws[p][16 + ml][kp];
        }
        acc0 = __builtin_amdgcn_wmma_f32_16x16x32_f16(false, a.v, false, b0.v,
                                                      (short)0, acc0, false, false);
        acc1 = __builtin_amdgcn_wmma_f32_16x16x32_f16(false, a.v, false, b1.v,
                                                      (short)0, acc1, false, false);
    }
    int col0 = n0 + ml, col1 = n0 + 16 + ml;
    float bv0 = bias[col0], bv1 = bias[col1];
#pragma unroll
    for (int v = 0; v < 8; ++v) {
        int row = m0 + wave * 16 + v + 8 * hh;
        float x0 = acc0[v] + bv0;
        float x1 = acc1[v] + bv1;
        C[(size_t)row * N + col0] = x0 > 0.f ? x0 : 0.f;
        C[(size_t)row * N + col1] = x1 > 0.f ? x1 : 0.f;
    }
}

// ---------------------------------------------------------------------------
// Transposed conv (stride 2, pad 2, 4x4 kernel) as phase-major implicit GEMM.
// Per phase (oy%2, ox%2) each output pixel uses a fixed 2x2 tap subset:
//   M = BATCH*IH*IW pixels (mult of 64), K = Ci*4, N = Co (mult of 32).
// Tap pairs (t, t+1) with even t share y and channel -> one row base per pair.
// Boundary handling is branchless (clamped address + value select).
// ---------------------------------------------------------------------------
__global__ void deconv_relu_wmma(const float* __restrict__ in,   // (B,Ci,IH,IW)
                                 const float* __restrict__ w,    // (Co,Ci,4,4)
                                 const float* __restrict__ bias, // (Co)
                                 float* __restrict__ out,        // (B,Co,2IH,2IW)
                                 int Ci, int Co, int IH, int IW)
{
    __shared__ unsigned As[2][64][16];
    __shared__ unsigned Ws[2][32][16];
    const int tid  = threadIdx.x;
    const int lane = tid & 31;
    const int wave = tid >> 5;
    const int hh   = (lane >> 4) & 1;
    const int ml   = lane & 15;
    const int py   = blockIdx.z >> 1, px = blockIdx.z & 1;
    const int m0   = blockIdx.y * 64;
    const int n0   = blockIdx.x * 32;
    const int HW   = IH * IW;
    const int OH   = 2 * IH, OW = 2 * IW;

    v8f acc0 = {}, acc1 = {};
    const int nsteps = Ci >> 3;          // 32 k-values = 8 channels x 4 taps
    for (int s = 0; s < nsteps; ++s) {
        const int k0 = s << 5;
        const int p  = s & 1;
        // A tile: implicit im2col gather, packed pairs (t even, t+1).
#pragma unroll
        for (int it = 0; it < 8; ++it) {
            int idx = tid + it * 128;
            int r = idx >> 4, cp = idx & 15;
            int gr  = m0 + r;
            int n   = gr / HW;
            int rem = gr - n * HW;
            int ry  = rem / IW, rx = rem - ry * IW;
            int k   = k0 + 2 * cp;
            int ci  = k >> 2;
            int t0  = k & 3;                       // 0 or 2
            int yy  = ry + py + (t0 >> 1) - 1;
            int x0  = rx + px - 1;
            int x1  = rx + px;
            int yc  = yy < 0 ? 0 : (yy > IH - 1 ? IH - 1 : yy);
            int xc0 = x0 < 0 ? 0 : x0;
            int xc1 = x1 > IW - 1 ? IW - 1 : x1;
            const float* base = in + ((size_t)(n * Ci + ci) * IH + yc) * IW;
            float f0 = base[xc0];
            float f1 = base[xc1];
            bool oky = (yy >= 0) & (yy < IH);
            if (!(oky & (x0 >= 0))) f0 = 0.f;
            if (!(oky & (x1 < IW))) f1 = 0.f;
            As[p][r][cp] = pack2h(f0, f1);
        }
        // B tile: tap-pair weights, [col][kpair].
#pragma unroll
        for (int it = 0; it < 4; ++it) {
            int idx = tid + it * 128;
            int cp = idx >> 5, n = idx & 31;
            int k  = k0 + 2 * cp;
            int ci = k >> 2;
            int t0 = k & 3;
            int ky = py + 2 * (t0 >> 1);
            int o  = n0 + n;
            const float* src = w + ((size_t)(o * Ci) + ci) * 16 + ky * 4 + px;
            Ws[p][n][cp] = pack2h(src[0], src[2]);    // kx = px, px+2
        }
        __syncthreads();
        Frag a, b0, b1;
#pragma unroll
        for (int j = 0; j < 8; ++j) {
            int kp = ((j < 4) ? j : (8 + (j - 4))) + hh * 4;
            a.u[j]  = As[p][wave * 16 + ml][kp];
            b0.u[j] = Ws[p][ml][kp];
            b1.u[j] = Ws[p][16 + ml][kp];
        }
        acc0 = __builtin_amdgcn_wmma_f32_16x16x32_f16(false, a.v, false, b0.v,
                                                      (short)0, acc0, false, false);
        acc1 = __builtin_amdgcn_wmma_f32_16x16x32_f16(false, a.v, false, b1.v,
                                                      (short)0, acc1, false, false);
    }
    int o0 = n0 + ml, o1 = n0 + 16 + ml;
    float bv0 = bias[o0], bv1 = bias[o1];
#pragma unroll
    for (int v = 0; v < 8; ++v) {
        int gr  = m0 + wave * 16 + v + 8 * hh;
        int n   = gr / HW;
        int rem = gr - n * HW;
        int ry  = rem / IW, rx = rem - ry * IW;
        int oy  = 2 * ry + py, ox = 2 * rx + px;
        float val0 = acc0[v] + bv0;
        float val1 = acc1[v] + bv1;
        out[((size_t)(n * Co + o0) * OH + oy) * OW + ox] = val0 > 0.f ? val0 : 0.f;
        out[((size_t)(n * Co + o1) * OH + oy) * OW + ox] = val1 > 0.f ? val1 : 0.f;
    }
}

// ---------------------------------------------------------------------------
// Final deconv (Co=1) fused with sigmoid -> field grid (B,64,64)
// ---------------------------------------------------------------------------
__global__ void deconv4_sigmoid(const float* __restrict__ in,   // (B,32,32,32)
                                const float* __restrict__ w,    // (1,32,4,4)
                                const float* __restrict__ bias,
                                float* __restrict__ field)      // (B,64,64)
{
    int gid = blockIdx.x * blockDim.x + threadIdx.x;
    if (gid >= BATCH * GRID * GRID) return;
    int n   = gid / (GRID * GRID);
    int rem = gid - n * (GRID * GRID);
    int oy = rem >> 6, ox = rem & 63;
    int py = oy & 1, px = ox & 1, ry = oy >> 1, rx = ox >> 1;
    float acc = bias[0];
    for (int i = 0; i < 32; ++i) {
#pragma unroll
        for (int t = 0; t < 4; ++t) {
            int y = ry + py + (t >> 1) - 1;
            int x = rx + px + (t & 1) - 1;
            int yc = y < 0 ? 0 : (y > 31 ? 31 : y);
            int xc = x < 0 ? 0 : (x > 31 ? 31 : x);
            float f = in[((size_t)(n * 32 + i) * 32 + yc) * 32 + xc];
            if (!(y >= 0 && y < 32 && x >= 0 && x < 32)) f = 0.f;
            int ky = py + 2 * (t >> 1), kx = px + 2 * (t & 1);
            acc += w[i * 16 + ky * 4 + kx] * f;
        }
    }
    field[gid] = 1.f / (1.f + expf(-acc));
}

// ---------------------------------------------------------------------------
// 9-tap separable gaussian blur, reflect padding (excl. edge)
// ---------------------------------------------------------------------------
__device__ __forceinline__ int refl(int i) {
    return i < 0 ? -i : (i > GRID - 1 ? 2 * (GRID - 1) - i : i);
}

__global__ void blur_v(const float* __restrict__ f, float* __restrict__ o)
{
    int gid = blockIdx.x * blockDim.x + threadIdx.x;
    if (gid >= BATCH * GRID * GRID) return;
    int n = gid / (GRID * GRID), rem = gid - n * (GRID * GRID);
    int y = rem >> 6, x = rem & 63;
    const float w[5] = {0.39894228f, 0.24197072f, 0.05399097f, 0.00443185f, 1.338302e-4f};
    const float* base = f + (size_t)n * GRID * GRID;
    float s = w[0] * base[y * GRID + x];
#pragma unroll
    for (int d = 1; d <= 4; ++d)
        s += w[d] * (base[refl(y - d) * GRID + x] + base[refl(y + d) * GRID + x]);
    o[gid] = s;
}

__global__ void blur_h(const float* __restrict__ f, float* __restrict__ o)
{
    int gid = blockIdx.x * blockDim.x + threadIdx.x;
    if (gid >= BATCH * GRID * GRID) return;
    int n = gid / (GRID * GRID), rem = gid - n * (GRID * GRID);
    int y = rem >> 6, x = rem & 63;
    const float w[5] = {0.39894228f, 0.24197072f, 0.05399097f, 0.00443185f, 1.338302e-4f};
    const float* base = f + (size_t)n * GRID * GRID;
    float s = w[0] * base[y * GRID + x];
#pragma unroll
    for (int d = 1; d <= 4; ++d)
        s += w[d] * (base[y * GRID + refl(x - d)] + base[y * GRID + refl(x + d)]);
    o[gid] = s;
}

// ---------------------------------------------------------------------------
// Peaks (5x5 max-filter) + top-K waypoints with fixed fallback ring
// ---------------------------------------------------------------------------
__global__ void waypoints_kernel(const float* __restrict__ sm, float* __restrict__ wps)
{
    __shared__ float ssm[GRID * GRID];
    __shared__ float ssc[GRID * GRID];
    __shared__ float rv[256];
    __shared__ int   ri[256];
    int b = blockIdx.x, tid = threadIdx.x;
    for (int p = tid; p < GRID * GRID; p += 256) ssm[p] = sm[(size_t)b * GRID * GRID + p];
    __syncthreads();
    for (int p = tid; p < GRID * GRID; p += 256) {
        int y = p >> 6, x = p & 63;
        float m = -1e30f;
        for (int dy = -2; dy <= 2; ++dy)
            for (int dx = -2; dx <= 2; ++dx) {
                float v = ssm[refl(y + dy) * GRID + refl(x + dx)];
                m = v > m ? v : m;
            }
        float v = ssm[p];
        ssc[p] = (v == m && v > 0.3f) ? v : -1.f;
    }
    __syncthreads();
    for (int kk = 0; kk < KWP; ++kk) {
        float bv = -1e30f; int bi = 0;
        for (int p = tid; p < GRID * GRID; p += 256) {
            float v = ssc[p];
            if (v > bv) { bv = v; bi = p; }
        }
        rv[tid] = bv; ri[tid] = bi;
        __syncthreads();
        for (int s = 128; s > 0; s >>= 1) {
            if (tid < s) {
                if (rv[tid + s] > rv[tid] ||
                    (rv[tid + s] == rv[tid] && ri[tid + s] < ri[tid])) {
                    rv[tid] = rv[tid + s]; ri[tid] = ri[tid + s];
                }
            }
            __syncthreads();
        }
        if (tid == 0) {
            float val = rv[0]; int idx = ri[0];
            float wy, wx;
            if (val > 0.f) { wy = (float)(idx >> 6); wx = (float)(idx & 63); }
            else {
                float ang = 6.28318530718f * (float)kk / 5.f;
                int fy = 32 + (int)(10.f * cosf(ang));
                int fx = 32 + (int)(10.f * sinf(ang));
                fy = fy < 0 ? 0 : (fy > 63 ? 63 : fy);
                fx = fx < 0 ? 0 : (fx > 63 ? 63 : fx);
                wy = (float)fy; wx = (float)fx;
            }
            wps[(b * KWP + kk) * 2 + 0] = wy;
            wps[(b * KWP + kk) * 2 + 1] = wx;
            ssc[idx] = -2.f;
        }
        __syncthreads();
    }
}

// ---------------------------------------------------------------------------
// Per-sample path sampling, max-min selection, softmax + categorical
// ---------------------------------------------------------------------------
__device__ __forceinline__ unsigned pcg(unsigned x) {
    x = x * 747796405u + 2891336453u;
    unsigned w = ((x >> ((x >> 28) + 4u)) ^ x) * 277803737u;
    return (w >> 22) ^ w;
}
__device__ __forceinline__ float u01(unsigned h) {
    return ((float)(h >> 8) + 0.5f) * (1.f / 16777216.f);
}
__device__ __forceinline__ float gaussrand(unsigned s) {
    float u1 = u01(pcg(s));
    float u2 = u01(pcg(s ^ 0x9E3779B9u));
    return sqrtf(-2.f * logf(u1)) * cosf(6.28318530718f * u2);
}

__global__ void per_sample_kernel(const float* __restrict__ field,
                                  const float* __restrict__ wps,
                                  float* __restrict__ prior,
                                  float* __restrict__ trajs,
                                  float* __restrict__ weights)
{
    __shared__ float raw[NPATH][TSTEP][2];
    __shared__ float pth[NPATH][TSTEP][2];
    __shared__ float sc[NPATH];
    __shared__ float Dh[NPATH][NPATH];
    int b = blockIdx.x, tid = threadIdx.x;
    const float stds[3] = {0.f, 0.1f, 0.2f};
    for (int p = tid; p < NPATH * TSTEP; p += 64) {
        int pi = p / TSTEP, t = p - pi * TSTEP;
        int v = pi / KWP, kk = pi - v * KWP;
        float tt   = (float)t / (float)(TSTEP - 1);
        float bump = sinf(3.14159265359f * tt);
        float wy = wps[(b * KWP + kk) * 2 + 0];
        float wx = wps[(b * KWP + kk) * 2 + 1];
        float by = 32.f + tt * (wy - 32.f);
        float bx = 32.f + tt * (wx - 32.f);
        unsigned s0 = (unsigned)b * 977u ^ (unsigned)pi * 131071u ^ (unsigned)t * 8191u;
        float ny = gaussrand(pcg(s0 ^ 0x01234567u)) * stds[v] * 6.4f;
        float nx = gaussrand(pcg(s0 ^ 0x89ABCDEFu)) * stds[v] * 6.4f;
        float yv = by + ny * bump, xv = bx + nx * bump;
        yv = yv < 0.f ? 0.f : (yv > 63.f ? 63.f : yv);
        xv = xv < 0.f ? 0.f : (xv > 63.f ? 63.f : xv);
        raw[pi][t][0] = yv; raw[pi][t][1] = xv;
    }
    __syncthreads();
    for (int p = tid; p < NPATH * TSTEP; p += 64) {
        int pi = p / TSTEP, t = p - pi * TSTEP;
        int tm = t - 1 < 0 ? 0 : t - 1, tp = t + 1 > TSTEP - 1 ? TSTEP - 1 : t + 1;
        pth[pi][t][0] = (raw[pi][tm][0] + raw[pi][t][0] + raw[pi][tp][0]) * (1.f / 3.f);
        pth[pi][t][1] = (raw[pi][tm][1] + raw[pi][t][1] + raw[pi][tp][1]) * (1.f / 3.f);
    }
    __syncthreads();
    for (int pi = tid; pi < NPATH; pi += 64) {
        float s = 0.f;
        for (int t = 0; t < TSTEP; ++t) {
            int yi = (int)roundf(pth[pi][t][0]); yi = yi < 0 ? 0 : (yi > 63 ? 63 : yi);
            int xi = (int)roundf(pth[pi][t][1]); xi = xi < 0 ? 0 : (xi > 63 ? 63 : xi);
            s += field[(size_t)b * GRID * GRID + yi * GRID + xi];
        }
        sc[pi] = s / (float)TSTEP;
    }
    __syncthreads();
    for (int pr = tid; pr < NPATH * NPATH; pr += 64) {
        int i = pr / NPATH, j = pr - i * NPATH;
        float mx = 0.f;
        for (int t = 0; t < TSTEP; ++t) {
            float mn = 1e30f;
            float yi = pth[i][t][0], xi = pth[i][t][1];
            for (int u = 0; u < TSTEP; ++u) {
                float dy = yi - pth[j][u][0], dx = xi - pth[j][u][1];
                float d = sqrtf(dy * dy + dx * dx);
                mn = d < mn ? d : mn;
            }
            mx = mn > mx ? mn : mx;
        }
        Dh[i][j] = mx;
    }
    __syncthreads();
    if (tid == 0) {
        float mind[NPATH]; bool msk[NPATH]; int sel[MSEL]; float wsel[MSEL];
        int i0 = 0; float bs = sc[0];
        for (int i = 1; i < NPATH; ++i) if (sc[i] > bs) { bs = sc[i]; i0 = i; }
        sel[0] = i0;
        for (int j = 0; j < NPATH; ++j) {
            msk[j] = (j == i0);
            mind[j] = Dh[i0][j] > Dh[j][i0] ? Dh[i0][j] : Dh[j][i0];
        }
        for (int m = 1; m < MSEL; ++m) {
            int bi = 0; float bv = -1e30f;
            for (int j = 0; j < NPATH; ++j) {
                float v = msk[j] ? -1e30f : mind[j];
                if (v > bv) { bv = v; bi = j; }
            }
            sel[m] = bi; msk[bi] = true;
            for (int j = 0; j < NPATH; ++j) {
                float h = Dh[bi][j] > Dh[j][bi] ? Dh[bi][j] : Dh[j][bi];
                mind[j] = mind[j] < h ? mind[j] : h;
            }
        }
        float mxv = -1e30f;
        for (int m = 0; m < MSEL; ++m) {
            float v = sc[sel[m]] * 10.f;   // /TEMP
            wsel[m] = v; if (v > mxv) mxv = v;
        }
        float sum = 0.f;
        for (int m = 0; m < MSEL; ++m) { wsel[m] = expf(wsel[m] - mxv); sum += wsel[m]; }
        for (int m = 0; m < MSEL; ++m) wsel[m] /= sum;
        float u = u01(pcg(0xC0FFEEu ^ (unsigned)b * 2654435761u));
        int si = MSEL - 1; float csum = 0.f;
        for (int m = 0; m < MSEL; ++m) { csum += wsel[m]; if (u < csum) { si = m; break; } }
        for (int m = 0; m < MSEL; ++m) {
            weights[b * MSEL + m] = wsel[m];
            int pi = sel[m];
            for (int t = 0; t < TSTEP; ++t) {
                float ty = -5.f + pth[pi][t][0] * (10.f / 63.f);
                float tx = -5.f + pth[pi][t][1] * (10.f / 63.f);
                trajs[((size_t)(b * MSEL + m) * TSTEP + t) * 2 + 0] = ty;
                trajs[((size_t)(b * MSEL + m) * TSTEP + t) * 2 + 1] = tx;
                if (m == si) {
                    prior[((size_t)b * TSTEP + t) * 2 + 0] = ty;
                    prior[((size_t)b * TSTEP + t) * 2 + 1] = tx;
                }
            }
        }
    }
}

// ---------------------------------------------------------------------------
__global__ void concat_kernel(const float* __restrict__ a, const float* __restrict__ b,
                              float* __restrict__ o)
{
    int gid = blockIdx.x * blockDim.x + threadIdx.x;
    if (gid >= BATCH * 512) return;
    int n = gid >> 9, j = gid & 511;
    o[gid] = (j < 256) ? a[n * 256 + j] : b[n * 256 + (j - 256)];
}

// ---------------------------------------------------------------------------
extern "C" void kernel_launch(void* const* d_in, const int* in_sizes, int n_in,
                              void* d_out, int out_size, void* d_ws, size_t ws_size,
                              hipStream_t stream)
{
    (void)in_sizes; (void)n_in; (void)out_size; (void)ws_size;
    const float* z_pooled = (const float*)d_in[0];
    const float* z_c  = (const float*)d_in[1];
    const float* w_f1 = (const float*)d_in[2];
    const float* b_f1 = (const float*)d_in[3];
    const float* w_f2 = (const float*)d_in[4];
    const float* b_f2 = (const float*)d_in[5];
    const float* w_d  = (const float*)d_in[6];
    const float* b_d  = (const float*)d_in[7];
    const float* k1 = (const float*)d_in[8];  const float* b1 = (const float*)d_in[9];
    const float* k2 = (const float*)d_in[10]; const float* b2 = (const float*)d_in[11];
    const float* k3 = (const float*)d_in[12]; const float* b3 = (const float*)d_in[13];
    const float* k4 = (const float*)d_in[14]; const float* b4 = (const float*)d_in[15];

    float* out    = (float*)d_out;
    float* prior  = out;                                       // (B,20,2)
    float* fieldo = prior  + (size_t)BATCH * TSTEP * 2;        // (B,64,64)
    float* trajso = fieldo + (size_t)BATCH * GRID * GRID;      // (B,3,20,2)
    float* wghts  = trajso + (size_t)BATCH * MSEL * TSTEP * 2; // (B,3)

    float* ws   = (float*)d_ws;
    float* zcat = ws;  ws += (size_t)BATCH * 512;
    float* h1   = ws;  ws += (size_t)BATCH * HIDDEN;
    float* h2   = ws;  ws += (size_t)BATCH * HIDDEN;
    float* x0   = ws;  ws += (size_t)BATCH * HIDDEN * 16;   // (B,256,4,4)
    float* x1   = ws;  ws += (size_t)BATCH * 128 * 64;      // (B,128,8,8)
    float* x2   = ws;  ws += (size_t)BATCH * 64 * 256;      // (B,64,16,16)
    float* x3   = ws;  ws += (size_t)BATCH * 32 * 1024;     // (B,32,32,32)
    float* tb   = ws;  ws += (size_t)BATCH * GRID * GRID;
    float* smb  = ws;  ws += (size_t)BATCH * GRID * GRID;
    float* wpsb = ws;  ws += (size_t)BATCH * KWP * 2;

    const int NPIX = BATCH * GRID * GRID;

    concat_kernel<<<(BATCH * 512 + 255) / 256, 256, 0, stream>>>(z_pooled, z_c, zcat);
    gemm_relu_wmma<<<dim3(256 / 32, BATCH / 64), 128, 0, stream>>>(zcat, w_f1, b_f1, h1, BATCH, 256, 512);
    gemm_relu_wmma<<<dim3(256 / 32, BATCH / 64), 128, 0, stream>>>(h1, w_f2, b_f2, h2, BATCH, 256, 256);
    gemm_relu_wmma<<<dim3(4096 / 32, BATCH / 64), 128, 0, stream>>>(h2, w_d, b_d, x0, BATCH, 4096, 256);
    deconv_relu_wmma<<<dim3(128 / 32, BATCH * 16 / 64, 4), 128, 0, stream>>>(x0, k1, b1, x1, 256, 128, 4, 4);
    deconv_relu_wmma<<<dim3(64 / 32, BATCH * 64 / 64, 4), 128, 0, stream>>>(x1, k2, b2, x2, 128, 64, 8, 8);
    deconv_relu_wmma<<<dim3(32 / 32, BATCH * 256 / 64, 4), 128, 0, stream>>>(x2, k3, b3, x3, 64, 32, 16, 16);
    deconv4_sigmoid<<<(NPIX + 255) / 256, 256, 0, stream>>>(x3, k4, b4, fieldo);
    blur_v<<<(NPIX + 255) / 256, 256, 0, stream>>>(fieldo, tb);
    blur_h<<<(NPIX + 255) / 256, 256, 0, stream>>>(tb, smb);
    waypoints_kernel<<<BATCH, 256, 0, stream>>>(smb, wpsb);
    per_sample_kernel<<<BATCH, 64, 0, stream>>>(fieldo, wpsb, prior, trajso, wghts);
}
